// WindowAttention3D_84000970375763
// MI455X (gfx1250) — compile-verified, matrix operand order checked by reading, not run on hardware
//
#include <hip/hip_runtime.h>

typedef __bf16 bf16_t;
typedef __attribute__((ext_vector_type(16))) __bf16 v16bf;
typedef __attribute__((ext_vector_type(8)))  __bf16 v8bf;
typedef __attribute__((ext_vector_type(8)))  float  v8f;

__device__ __forceinline__ bf16_t f2bf(float f) { return (bf16_t)f; }

// Build a 16-elem bf16 WMMA fragment from two contiguous 8x bf16 (16B) LDS reads.
__device__ __forceinline__ v16bf make_frag(const bf16_t* p_lo, const bf16_t* p_hi) {
    v8bf lo = *(const v8bf*)p_lo;
    v8bf hi = *(const v8bf*)p_hi;
    return __builtin_shufflevector(lo, hi, 0,1,2,3,4,5,6,7,8,9,10,11,12,13,14,15);
}

// ---------------------------------------------------------------------------
// Precompute padded + masked relative-position bias: biasPad[h][m][n],
// h<32, m<112, n<128.  n>=98 -> -1e30 (mask), m>=98 -> 0 (padded q rows).
// Done ONCE instead of re-gathering in all 16384 attention blocks.
// ---------------------------------------------------------------------------
__global__ __launch_bounds__(256)
void build_bias_pad(const float* __restrict__ bias_table,
                    const int* __restrict__ rel_index,
                    float* __restrict__ biasPad)
{
    int idx = blockIdx.x * 256 + threadIdx.x;
    if (idx >= 32 * 112 * 128) return;
    int n = idx & 127;
    int m = (idx >> 7) % 112;
    int h = idx / (112 * 128);
    float v;
    if (n >= 98)      v = -1e30f;
    else if (m >= 98) v = 0.f;
    else              v = bias_table[(size_t)rel_index[m * 98 + n] * 32 + h];
    biasPad[idx] = v;
}

// ---------------------------------------------------------------------------
// Generic bf16-WMMA GEMM: out = epi(A[MxK] @ B[KxN] + bias[N])
// Block tile 64x128, K-step 32; 8 waves, each wave owns a 32x32 patch.
// EPI 0: out = acc + bias        (row-major out, ldo)
// EPI 1: out = tanh(acc + bias)
// EPI 2: qkv[m*3072 + 1024 + n] += 0.1f * (acc + bias)   (adapter K/V inject)
// ---------------------------------------------------------------------------
template<int EPI>
__global__ __launch_bounds__(256)
void gemm_bf16_wmma(const float* __restrict__ A, const float* __restrict__ B,
                    const float* __restrict__ bias, float* __restrict__ out,
                    int M, int N, int K, int lda, int ldb, int ldo,
                    float* __restrict__ qkv)
{
    constexpr int BM = 64, BN = 128, BK = 32;
    __shared__ __align__(16) bf16_t As[BM * BK];        // [m][k] row-major
    __shared__ __align__(16) bf16_t Bs[BN * BK];        // [n][k] (transposed)

    const int tid  = threadIdx.x;
    const int lane = tid & 31;
    const int wave = tid >> 5;
    const int wm   = wave & 1;          // 2 wave-rows  * 32
    const int wn   = wave >> 1;         // 4 wave-cols  * 32
    const int rowBase = blockIdx.y * BM;
    const int colBase = blockIdx.x * BN;

    const int laneRow = lane & 15;
    const int kb      = (lane >> 4) * 8;   // K sub-offset per ISA A/B layout
    const int mOff    = (lane >> 4) * 8;   // C/D row group

    v8f acc[2][2];
    #pragma unroll
    for (int i = 0; i < 2; ++i)
        #pragma unroll
        for (int j = 0; j < 2; ++j) acc[i][j] = {};

    for (int k0 = 0; k0 < K; k0 += BK) {
        // ---- stage A tile (64x32 fp32 -> bf16), 2 float4 per thread ----
        #pragma unroll
        for (int p = 0; p < 2; ++p) {
            int i  = tid + p * 256;
            int r  = i >> 3;
            int c4 = (i & 7) << 2;
            int gr = rowBase + r;
            float4 v = make_float4(0.f, 0.f, 0.f, 0.f);
            if (gr < M) {
                const float* ap = &A[(size_t)gr * lda + k0 + c4];
                v = *(const float4*)ap;
                if (k0 + BK < K) __builtin_prefetch(ap + BK, 0, 1);  // next K tile
            }
            As[r * BK + c4 + 0] = f2bf(v.x);
            As[r * BK + c4 + 1] = f2bf(v.y);
            As[r * BK + c4 + 2] = f2bf(v.z);
            As[r * BK + c4 + 3] = f2bf(v.w);
        }
        // ---- stage B tile (32x128 fp32 -> bf16 transposed), 4 float4 per thread ----
        #pragma unroll
        for (int p = 0; p < 4; ++p) {
            int i  = tid + p * 256;
            int kk = i >> 5;
            int n4 = (i & 31) << 2;
            int gn = colBase + n4;
            float4 v = make_float4(0.f, 0.f, 0.f, 0.f);
            if (gn + 3 < N) {
                const float* bp = &B[(size_t)(k0 + kk) * ldb + gn];
                v = *(const float4*)bp;
                if (k0 + BK < K) __builtin_prefetch(bp + (size_t)BK * ldb, 0, 1);
            }
            Bs[(n4 + 0) * BK + kk] = f2bf(v.x);
            Bs[(n4 + 1) * BK + kk] = f2bf(v.y);
            Bs[(n4 + 2) * BK + kk] = f2bf(v.z);
            Bs[(n4 + 3) * BK + kk] = f2bf(v.w);
        }
        __syncthreads();

        v16bf af[2], bfg[2];
        #pragma unroll
        for (int mi = 0; mi < 2; ++mi) {
            int r = wm * 32 + mi * 16 + laneRow;
            af[mi] = make_frag(&As[r * BK + kb], &As[r * BK + 16 + kb]);
        }
        #pragma unroll
        for (int ni = 0; ni < 2; ++ni) {
            int c = wn * 32 + ni * 16 + laneRow;
            bfg[ni] = make_frag(&Bs[c * BK + kb], &Bs[c * BK + 16 + kb]);
        }
        #pragma unroll
        for (int mi = 0; mi < 2; ++mi)
            #pragma unroll
            for (int ni = 0; ni < 2; ++ni)
                acc[mi][ni] = __builtin_amdgcn_wmma_f32_16x16x32_bf16(
                    false, af[mi], false, bfg[ni], (short)0, acc[mi][ni], false, false);
        __syncthreads();
    }

    // ---- epilogue ----
    #pragma unroll
    for (int mi = 0; mi < 2; ++mi)
        #pragma unroll
        for (int ni = 0; ni < 2; ++ni)
            #pragma unroll
            for (int j = 0; j < 8; ++j) {
                int m = rowBase + wm * 32 + mi * 16 + mOff + j;
                int n = colBase + wn * 32 + ni * 16 + laneRow;
                if (m < M && n < N) {
                    float v = acc[mi][ni][j] + bias[n];
                    if constexpr (EPI == 1) v = tanhf(v);
                    if constexpr (EPI == 2) {
                        qkv[(size_t)m * 3072 + 1024 + n] += 0.1f * v;
                    } else {
                        out[(size_t)m * ldo + n] = v;
                    }
                }
            }
}

// ---------------------------------------------------------------------------
// Fused window attention: one block per (b, h). N=98 tokens padded to 112
// rows (S) / 128 cols (P·V contraction). d=32 == WMMA K exactly.
// Bias+mask is applied from the precomputed biasPad during the softmax pass.
// ---------------------------------------------------------------------------
__global__ __launch_bounds__(256)
void attn_fused(const float* __restrict__ qkv, const float* __restrict__ biasPad,
                float* __restrict__ attOut)
{
    constexpr int N = 98, NP = 112, NC = 128, D = 32;
    __shared__ __align__(16) bf16_t qs [NP * D];    // [tok][dh], q pre-scaled
    __shared__ __align__(16) bf16_t ksh[NP * D];    // [tok][dh]
    __shared__ __align__(16) bf16_t vT [D * NC];    // [dh][tok], zero-padded
    __shared__ __align__(16) float  sBuf[NP * NC];  // logits; P (bf16) aliased per-row

    const int h = blockIdx.x, b = blockIdx.y;
    const int tid = threadIdx.x, lane = tid & 31, wave = tid >> 5;
    const int laneRow = lane & 15;
    const int kb      = (lane >> 4) * 8;
    const int mOff    = (lane >> 4) * 8;
    const float scale = 0.17677669529663687f;       // 32^-0.5
    const size_t base = (size_t)b * N * 3072 + (size_t)h * 32;
    const float* bh   = biasPad + (size_t)h * NP * NC;

    // ---- stage Q, K (112x32, pad zero) ----
    for (int idx = tid; idx < NP * D; idx += 256) {
        int t = idx >> 5, dh = idx & 31;
        float qv = 0.f, kv = 0.f;
        if (t < N) {
            qv = qkv[base + (size_t)t * 3072 + dh] * scale;
            kv = qkv[base + (size_t)t * 3072 + 1024 + dh];
        }
        qs[idx]  = f2bf(qv);
        ksh[idx] = f2bf(kv);
    }
    // ---- stage V transposed (32x128, pad zero) ----
    for (int idx = tid; idx < D * NC; idx += 256) {
        int dh = idx >> 7, t = idx & 127;
        float vv = (t < N) ? qkv[base + (size_t)t * 3072 + 2048 + dh] : 0.f;
        vT[dh * NC + t] = f2bf(vv);
    }
    // ---- init logits columns 112..127 (never written by S tiles) ----
    for (int idx = tid; idx < NP * 16; idx += 256) {
        int r = idx >> 4, c = 112 + (idx & 15);
        sBuf[r * NC + c] = 0.f;                      // biasPad adds -1e30 there
    }
    __syncthreads();

    // ---- S = (Q*scale) K^T : 7x7 tiles over 8 waves, raw scores to LDS ----
    for (int t49 = wave; t49 < 49; t49 += 8) {
        int mt = t49 / 7, nt = t49 % 7;
        int ar = mt * 16 + laneRow;
        v16bf a  = make_frag(&qs[ar * D + kb],  &qs[ar * D + 16 + kb]);
        int bc = nt * 16 + laneRow;
        v16bf bb = make_frag(&ksh[bc * D + kb], &ksh[bc * D + 16 + kb]);
        v8f s = {};
        s = __builtin_amdgcn_wmma_f32_16x16x32_bf16(false, a, false, bb, (short)0, s, false, false);
        #pragma unroll
        for (int j = 0; j < 8; ++j)
            sBuf[(mt * 16 + mOff + j) * NC + nt * 16 + laneRow] = s[j];
    }
    __syncthreads();

    // ---- row softmax with fused bias/mask add (coalesced, L2-resident) ----
    bf16_t* pBuf = (bf16_t*)sBuf;                   // P row r at pBuf[r*256 + c]
    for (int r = wave; r < NP; r += 8) {
        float v4[4];
        float mx = -1e30f;
        #pragma unroll
        for (int i = 0; i < 4; ++i) {
            int c = lane + 32 * i;
            v4[i] = sBuf[r * NC + c] + bh[r * NC + c];
            mx = fmaxf(mx, v4[i]);
        }
        #pragma unroll
        for (int o = 16; o > 0; o >>= 1) mx = fmaxf(mx, __shfl_xor(mx, o, 32));
        float sum = 0.f;
        #pragma unroll
        for (int i = 0; i < 4; ++i) { v4[i] = __expf(v4[i] - mx); sum += v4[i]; }
        #pragma unroll
        for (int o = 16; o > 0; o >>= 1) sum += __shfl_xor(sum, o, 32);
        float inv = 1.f / sum;
        #pragma unroll
        for (int i = 0; i < 4; ++i) pBuf[r * 256 + lane + 32 * i] = f2bf(v4[i] * inv);
    }
    __syncthreads();

    // ---- O = P @ V : 7x2 output tiles, K = 128 in 4 WMMA steps ----
    for (int t14 = wave; t14 < 14; t14 += 8) {
        int mt = t14 >> 1, nt = t14 & 1;
        v8f o = {};
        #pragma unroll
        for (int ks = 0; ks < 4; ++ks) {
            int ar = mt * 16 + laneRow;
            v16bf a  = make_frag(&pBuf[ar * 256 + ks * 32 + kb],
                                 &pBuf[ar * 256 + ks * 32 + 16 + kb]);
            int bc = nt * 16 + laneRow;
            v16bf bb = make_frag(&vT[bc * NC + ks * 32 + kb],
                                 &vT[bc * NC + ks * 32 + 16 + kb]);
            o = __builtin_amdgcn_wmma_f32_16x16x32_bf16(false, a, false, bb, (short)0, o, false, false);
        }
        #pragma unroll
        for (int j = 0; j < 8; ++j) {
            int m  = mt * 16 + mOff + j;
            int dh = nt * 16 + laneRow;
            if (m < N)
                attOut[((size_t)b * N + m) * 1024 + (size_t)h * 32 + dh] = o[j];
        }
    }
}

extern "C" void kernel_launch(void* const* d_in, const int* in_sizes, int n_in,
                              void* d_out, int out_size, void* d_ws, size_t ws_size,
                              hipStream_t stream)
{
    (void)in_sizes; (void)n_in; (void)out_size; (void)ws_size;
    const float* x          = (const float*)d_in[0];
    const float* w_qkv      = (const float*)d_in[1];
    const float* b_qkv      = (const float*)d_in[2];
    const float* w_a1       = (const float*)d_in[3];
    const float* b_a1       = (const float*)d_in[4];
    const float* w_a2       = (const float*)d_in[5];
    const float* b_a2       = (const float*)d_in[6];
    const float* bias_table = (const float*)d_in[7];
    const float* w_proj     = (const float*)d_in[8];
    const float* b_proj     = (const float*)d_in[9];
    const int*   rel_index  = (const int*)d_in[10];

    const int B = 512, N = 98, C = 1024;
    const int M = B * N;                                  // 50176, divisible by 64

    float* qkv     = (float*)d_ws;                        // M x 3072
    float* a1      = qkv + (size_t)M * 3072;              // M x 64
    float* att     = a1  + (size_t)M * 64;                // M x 1024
    float* biasPad = att + (size_t)M * 1024;              // 32 x 112 x 128

    dim3 blk(256);
    const int mBlocks = (M + 63) / 64;

    // 0) padded/masked relative-position bias (h-only, shared by all 512 windows)
    build_bias_pad<<<(32 * 112 * 128 + 255) / 256, blk, 0, stream>>>(
        bias_table, rel_index, biasPad);
    // 1) qkv = x @ w_qkv + b_qkv
    gemm_bf16_wmma<0><<<dim3(3072 / 128, mBlocks), blk, 0, stream>>>(
        x, w_qkv, b_qkv, qkv, M, 3 * C, C, C, 3 * C, 3 * C, nullptr);
    // 2) a1 = tanh(x @ w_a1 + b_a1)
    gemm_bf16_wmma<1><<<dim3(1, mBlocks), blk, 0, stream>>>(
        x, w_a1, b_a1, a1, M, 64, C, C, 64, 64, nullptr);
    // 3) qkv[:,1024:3072] += 0.1 * (a1 @ w_a2 + b_a2)
    gemm_bf16_wmma<2><<<dim3(2048 / 128, mBlocks), blk, 0, stream>>>(
        a1, w_a2, b_a2, nullptr, M, 2 * C, 64, 64, 2 * C, 0, qkv);
    // 4) fused attention per (b, h)
    attn_fused<<<dim3(32, 512), blk, 0, stream>>>(qkv, biasPad, att);
    // 5) out = att @ w_proj + b_proj
    gemm_bf16_wmma<0><<<dim3(1024 / 128, mBlocks), blk, 0, stream>>>(
        att, w_proj, b_proj, (float*)d_out, M, C, C, C, C, C, nullptr);
}
